// GanaRGCNConv_27522150433361
// MI455X (gfx1250) — compile-verified, matrix operand order checked by reading, not run on hardware
//
#include <hip/hip_runtime.h>
#include <math.h>

// ---- problem constants (match reference) ----
#define NNODES 50000            // multiple of 16 -> every 16-row tile is full
#define NEDGES 800000
#define NREL   6
#define NBASES 30
#define NSLOT  (NREL + 1)       // 6 relations + root folded in as slot 6

typedef __attribute__((ext_vector_type(16))) _Float16 v16h;
typedef __attribute__((ext_vector_type(8)))  _Float16 v8h;
typedef __attribute__((ext_vector_type(8)))  float    v8f;

// ---------------------------------------------------------------------------
// small utility kernels
// ---------------------------------------------------------------------------
__global__ void fill_f32(float* __restrict__ p, float v, int n) {
  int i = blockIdx.x * blockDim.x + threadIdx.x;
  if (i < n) p[i] = v;
}

__global__ void fill_i32(int* __restrict__ p, int v, int n) {
  int i = blockIdx.x * blockDim.x + threadIdx.x;
  if (i < n) p[i] = v;
}

__global__ void cvt_f32_to_f16(const float* __restrict__ x,
                               _Float16* __restrict__ y, int n) {
  int i = blockIdx.x * blockDim.x + threadIdx.x;
  if (i < n) y[i] = (_Float16)x[i];
}

// per-(relation,dst) edge counts -- identical for all 3 layers, computed once
__global__ void count_edges(const int* __restrict__ etype,
                            const int* __restrict__ dst,
                            int* __restrict__ cnt) {
  int e = blockIdx.x * blockDim.x + threadIdx.x;
  if (e < NEDGES) atomicAdd(&cnt[etype[e] * NNODES + dst[e]], 1);
}

__global__ void inv_counts(const int* __restrict__ cnt,
                           float* __restrict__ inv, int n) {
  int i = blockIdx.x * blockDim.x + threadIdx.x;
  if (i < n) inv[i] = 1.0f / (float)(cnt[i] > 1 ? cnt[i] : 1);
}

// ---------------------------------------------------------------------------
// Weight build + pack:  W[slot] = sum_b comp[slot,b]*bases[b]  (slot<NREL),
//                       W[NREL] = root.   Packed f16 in B-fragment lane-major
// order for v_wmma_f32_16x16x32_f16:
//   fragment fid = (slot*cgs + cg)*2 + ks   (ks = k-step of 32)
//   element (lane l, j):  n = l&15, hi = l>>4
//     j<8 :  k = ks*32 + hi*8 + j
//     j>=8:  k = ks*32 + 16 + hi*8 + (j-8)
// so the GEMM reads B with one contiguous 32B load per lane.
// ---------------------------------------------------------------------------
__global__ void pack_w(const float* __restrict__ bases,
                       const float* __restrict__ comp,
                       const float* __restrict__ root,
                       _Float16* __restrict__ wpk,
                       int din, int dout, int out_pad) {
  const int cgs = out_pad >> 4;
  const int total = NSLOT * cgs * 2 * 512;
  int i = blockIdx.x * blockDim.x + threadIdx.x;
  if (i >= total) return;
  int fid = i >> 9;
  int lj  = i & 511;
  int l   = lj >> 4;
  int j   = lj & 15;
  int ks  = fid & 1;
  int t   = fid >> 1;
  int slot = t / cgs, cg = t % cgs;
  int n  = l & 15, hi = l >> 4;
  int k  = ks * 32 + (j < 8 ? hi * 8 + j : 16 + hi * 8 + (j - 8));
  int nc = cg * 16 + n;
  float val = 0.0f;
  if (nc < dout && k < din) {
    if (slot < NREL) {
      for (int b = 0; b < NBASES; ++b)
        val += comp[slot * NBASES + b] * bases[(b * din + k) * dout + nc];
    } else {
      val = root[k * dout + nc];
    }
  }
  wpk[i] = (_Float16)val;
}

// ---------------------------------------------------------------------------
// WMMA GEMM:  hrel[slot][n][OUT_PAD] = h16[n][0:64] @ W[slot]  for all slots.
// One wave = one full 16-row strip (NNODES % 16 == 0, so no predication and
// EXEC is all-ones at every WMMA).  K=64 -> two v_wmma_f32_16x16x32_f16 per
// 16-col tile, f32 accumulate, unconditional f16 stores.
// CGS = column groups of 16 per slot (compile-time: 4 or 1).
// ---------------------------------------------------------------------------
template <int CGS>
__global__ void rgcn_gemm(const _Float16* __restrict__ h16,
                          const _Float16* __restrict__ wpk,
                          _Float16* __restrict__ hrel,
                          int nrows) {
  constexpr int OP = CGS * 16;             // padded output stride
  const int lane = threadIdx.x & 31;
  const int wave = threadIdx.x >> 5;
  const int m0 = blockIdx.x * 128 + wave * 16;
  if (m0 >= nrows) return;                 // wave-uniform exit

  const int half = lane & 15;
  const int hi   = lane >> 4;
  const _Float16* rp = h16 + (size_t)(m0 + half) * 64;

  // A fragments per ISA 16-bit A layout (two k-steps of 32)
  v16h a0, a1;
  {
    v8h l0 = *(const v8h*)(rp + hi * 8);
    v8h h0 = *(const v8h*)(rp + 16 + hi * 8);
    v8h l1 = *(const v8h*)(rp + 32 + hi * 8);
    v8h h1 = *(const v8h*)(rp + 48 + hi * 8);
#pragma unroll
    for (int i = 0; i < 8; ++i) {
      a0[i] = l0[i]; a0[i + 8] = h0[i];
      a1[i] = l1[i]; a1[i + 8] = h1[i];
    }
  }

  const v16h* bp = (const v16h*)wpk;       // 32B per lane, contiguous
  for (int slot = 0; slot < NSLOT; ++slot) {
    // row base for this lane's C rows: lanes 0-15 -> m0..m0+7 (v=0..7),
    // lanes 16-31 -> m0+8..m0+15
    _Float16* sp = hrel + (size_t)slot * nrows * OP
                        + (size_t)(m0 + hi * 8) * OP + half;
#pragma unroll
    for (int cg = 0; cg < CGS; ++cg) {
      const int fb = (slot * CGS + cg) * 2;
      v16h b0 = bp[(fb + 0) * 32 + lane];
      v16h b1 = bp[(fb + 1) * 32 + lane];
      v8f c = {};
      c = __builtin_amdgcn_wmma_f32_16x16x32_f16(false, a0, false, b0,
                                                 (short)0, c, false, false);
      c = __builtin_amdgcn_wmma_f32_16x16x32_f16(false, a1, false, b1,
                                                 (short)0, c, false, false);
      _Float16* op = sp + cg * 16;
#pragma unroll
      for (int v = 0; v < 8; ++v)
        op[(size_t)v * OP] = (_Float16)c[v];   // unconditional, coalesced b16
    }
  }
}

// ---------------------------------------------------------------------------
// Edge message scatter: agg[dst,c] += hrel[etype][src][c] / cnt(etype,dst)
// (pre-scaling by 1/cnt lets mean-per-relation collapse into ONE [N,OC]
//  atomic accumulator instead of [R*N,OC]).  OC/OUT_PAD are powers of two,
//  compile-time -> index math is shifts.
// ---------------------------------------------------------------------------
template <int OC, int OUT_PAD>
__global__ void rgcn_scatter(const _Float16* __restrict__ hrel,
                             const float* __restrict__ invcnt,
                             const int* __restrict__ src,
                             const int* __restrict__ dst,
                             const int* __restrict__ etype,
                             float* __restrict__ agg) {
  long long i = (long long)blockIdx.x * blockDim.x + threadIdx.x;
  if (i >= (long long)NEDGES * OC) return;
  int e = (int)(i / OC);
  int c = (int)(i % OC);
  int t = etype[e], s = src[e], d = dst[e];
  float v = (float)hrel[(size_t)t * NNODES * OUT_PAD + (size_t)s * OUT_PAD + c];
  v *= invcnt[(size_t)t * NNODES + d];
  atomicAdd(&agg[(size_t)d * OC + c], v);
}

// out = relu(agg + root_part + bias), written f16 for the next layer's GEMM
__global__ void combine_relu(const float* __restrict__ agg,
                             const _Float16* __restrict__ hroot,
                             const float* __restrict__ bias,
                             _Float16* __restrict__ hnext, int n) {
  int i = blockIdx.x * blockDim.x + threadIdx.x;
  if (i >= n) return;
  int c = i & 63;
  float v = agg[i] + (float)hroot[i] + bias[c];   // hroot stride == 64 == OC
  hnext[i] = (_Float16)fmaxf(v, 0.0f);
}

// final layer: x1 = agg + root_part + bias  (C=2), output log_softmax(x1)
__global__ void combine_logsoftmax(const float* __restrict__ agg,
                                   const _Float16* __restrict__ hroot,
                                   const float* __restrict__ bias,
                                   float* __restrict__ out) {
  int n = blockIdx.x * blockDim.x + threadIdx.x;
  if (n >= NNODES) return;
  float r0 = agg[n * 2 + 0] + (float)hroot[(size_t)n * 16 + 0] + bias[0];
  float r1 = agg[n * 2 + 1] + (float)hroot[(size_t)n * 16 + 1] + bias[1];
  float m = fmaxf(r0, r1);
  float lse = m + logf(expf(r0 - m) + expf(r1 - m));
  out[n * 2 + 0] = r0 - lse;
  out[n * 2 + 1] = r1 - lse;
}

// ---------------------------------------------------------------------------
extern "C" void kernel_launch(void* const* d_in, const int* in_sizes, int n_in,
                              void* d_out, int out_size, void* d_ws, size_t ws_size,
                              hipStream_t stream) {
  const float* x      = (const float*)d_in[0];
  const int*   eidx   = (const int*)d_in[1];
  const int*   etype  = (const int*)d_in[2];
  const int*   src    = eidx;
  const int*   dst    = eidx + NEDGES;

  const float* bases[3] = { (const float*)d_in[3], (const float*)d_in[7],  (const float*)d_in[11] };
  const float* comp [3] = { (const float*)d_in[4], (const float*)d_in[8],  (const float*)d_in[12] };
  const float* root [3] = { (const float*)d_in[5], (const float*)d_in[9],  (const float*)d_in[13] };
  const float* bias [3] = { (const float*)d_in[6], (const float*)d_in[10], (const float*)d_in[14] };

  // ---- workspace carve (all offsets 256B aligned) ----
  char* ws = (char*)d_ws;
  size_t off = 0;
  auto carve = [&](size_t bytes) { void* p = ws + off; off += (bytes + 255) & ~(size_t)255; return p; };
  _Float16* h16    = (_Float16*)carve((size_t)NNODES * 64 * 2);          //  6.4 MB
  _Float16* hrel   = (_Float16*)carve((size_t)NSLOT * NNODES * 64 * 2);  // 44.8 MB
  float*    agg    = (float*)   carve((size_t)NNODES * 64 * 4);          // 12.8 MB
  int*      cnt    = (int*)     carve((size_t)NREL * NNODES * 4);        //  1.2 MB
  float*    invc   = (float*)   carve((size_t)NREL * NNODES * 4);        //  1.2 MB
  _Float16* wpk    = (_Float16*)carve((size_t)NSLOT * 64 * 64 * 2);      //  57 KB
  (void)ws_size; (void)n_in; (void)in_sizes; (void)out_size;

  const int T = 256;
  auto nb = [](long long n, int t) { return (int)((n + t - 1) / t); };

  // ---- one-time: f16 features + per-(rel,dst) mean reciprocals ----
  cvt_f32_to_f16<<<nb((long long)NNODES * 64, T), T, 0, stream>>>(x, h16, NNODES * 64);
  fill_i32<<<nb((long long)NREL * NNODES, T), T, 0, stream>>>(cnt, 0, NREL * NNODES);
  count_edges<<<nb(NEDGES, T), T, 0, stream>>>(etype, dst, cnt);
  inv_counts<<<nb((long long)NREL * NNODES, T), T, 0, stream>>>(cnt, invc, NREL * NNODES);

  const int gemm_blocks = (NNODES + 127) / 128;

  // ---- layers 0 and 1: 64 -> 64 ----
  for (int l = 0; l < 2; ++l) {
    pack_w<<<nb((long long)NSLOT * 4 * 2 * 512, T), T, 0, stream>>>(
        bases[l], comp[l], root[l], wpk, 64, 64, 64);
    rgcn_gemm<4><<<gemm_blocks, T, 0, stream>>>(h16, wpk, hrel, NNODES);
    fill_f32<<<nb((long long)NNODES * 64, T), T, 0, stream>>>(agg, 0.0f, NNODES * 64);
    rgcn_scatter<64, 64><<<nb((long long)NEDGES * 64, T), T, 0, stream>>>(
        hrel, invc, src, dst, etype, agg);
    combine_relu<<<nb((long long)NNODES * 64, T), T, 0, stream>>>(
        agg, hrel + (size_t)NREL * NNODES * 64, bias[l], h16, NNODES * 64);
  }

  // ---- layer 2: 64 -> 2 (padded to one 16-column tile) ----
  pack_w<<<nb((long long)NSLOT * 1 * 2 * 512, T), T, 0, stream>>>(
      bases[2], comp[2], root[2], wpk, 64, 2, 16);
  rgcn_gemm<1><<<gemm_blocks, T, 0, stream>>>(h16, wpk, hrel, NNODES);
  fill_f32<<<nb((long long)NNODES * 2, T), T, 0, stream>>>(agg, 0.0f, NNODES * 2);
  rgcn_scatter<2, 16><<<nb((long long)NEDGES * 2, T), T, 0, stream>>>(
      hrel, invc, src, dst, etype, agg);
  combine_logsoftmax<<<nb(NNODES, T), T, 0, stream>>>(
      agg, hrel + (size_t)NREL * NNODES * 16, bias[2], (float*)d_out);
}
